// MyModel_48395691491912
// MI455X (gfx1250) — matrix operand order repaired and checked by reading, not compiled, not run on hardware
//
#include <hip/hip_runtime.h>

// ---------------------------------------------------------------------------
// Tacotron-style encoder LSTM + location-sensitive-attention decoder, CDNA5.
// One block = 16 batch rows (WMMA M), 5 wave32s = 5 N-tiles of the 80 gates.
// Gate GEMMs: v_wmma_f32_16x16x32_bf16, weight B-fragments resident in VGPRs.
// Latency-path tuning: rcp-based sigmoid/tanh, parallel softmax with folded
// normalization, double-buffered x_t prefetch.
// ---------------------------------------------------------------------------

typedef __attribute__((ext_vector_type(16))) __bf16 v16bf;
typedef __attribute__((ext_vector_type(8)))  float  v8f;

#define BATCH   512
#define FEAT    80
#define LEN     111
#define TSTEPS  222
#define UNITS   20
#define ATTN    10
#define CFILT   32
#define GATES   80            // 4*UNITS
#define TILE    16            // batch rows per block (WMMA M)
#define NWAVES  5             // one wave per 16-gate N-tile
#define NTHREADS (NWAVES * 32)
#define NBTILES (BATCH / TILE)
#define XPREF   ((TILE * FEAT) / NTHREADS)   // 8 dwords/thread

// fast nonlinearities: bf16 matmul inputs make IEEE-divide precision wasted;
// v_rcp_f32 keeps the pointwise chain short (serial-latency-bound workload).
__device__ __forceinline__ float sigf(float x) {
    return __builtin_amdgcn_rcpf(1.0f + __expf(-x));
}
__device__ __forceinline__ float tanh_fast(float x) {
    // 1 - 2/(e^{2x}+1); exp overflow/underflow saturates to +/-1 correctly
    return 1.0f - 2.0f * __builtin_amdgcn_rcpf(__expf(2.0f * x) + 1.0f);
}

__device__ __forceinline__ v8f wmma_bf16(v16bf a, v16bf b, v8f c) {
    return __builtin_amdgcn_wmma_f32_16x16x32_bf16(false, a, false, b, (short)0, c,
                                                   false, false);
}

// B-fragment (32x16) of row-major f32 W[K][ldn], zero-padded outside [K,N).
// lanes 0-15 hold K=0..15 of column n, lanes 16-31 hold K=16..31; elem j -> K=j.
__device__ __forceinline__ v16bf loadB_g(const float* W, int ldn, int kbase,
                                         int nbase, int K, int N) {
    int ln = threadIdx.x & 31;
    int n  = nbase + (ln & 15);
    int k0 = kbase + ((ln >> 4) << 4);
    v16bf r;
#pragma unroll
    for (int j = 0; j < 16; ++j) {
        int   k = k0 + j;
        float v = (k < K && n < N) ? W[k * ldn + n] : 0.0f;
        r[j] = (__bf16)v;
    }
    return r;
}

// A-fragment (16x32) from LDS row-major f32 X[16][ld], zero-padded past K.
// lane m holds K=0..15, lane m+16 K=16..31; j -> j+half*16.
__device__ __forceinline__ v16bf loadA_s(const float* X, int ld, int kbase, int K) {
    int ln   = threadIdx.x & 31;
    int m    = ln & 15;
    int half = ln >> 4;
    v16bf r;
#pragma unroll
    for (int j = 0; j < 16; ++j) {
        int   k = kbase + half * 16 + j;
        float v = (k < K) ? X[m * ld + k] : 0.0f;
        r[j] = (__bf16)v;
    }
    return r;
}

__global__ __launch_bounds__(NTHREADS) void taco_kernel(
    const float* __restrict__ inputs,   // [B,L,F]
    const float* __restrict__ enc_Wx,   // [F,4U]
    const float* __restrict__ enc_Wh,   // [U,4U]
    const float* __restrict__ enc_b,    // [4U]
    const float* __restrict__ att_Wq,   // [U,A]
    const float* __restrict__ att_bq,   // [A]
    const float* __restrict__ att_Wm,   // [U,A]
    const float* __restrict__ att_bm,   // [A]
    const float* __restrict__ att_V,    // [A,1]
    const float* __restrict__ att_bv,   // [1]
    const float* __restrict__ loc_Wconv,// [1,CF]
    const float* __restrict__ loc_Wd,   // [CF,A]
    const float* __restrict__ dec_Wx,   // [U,4U]
    const float* __restrict__ dec_Wh,   // [U,4U]
    const float* __restrict__ dec_b,    // [4U]
    const float* __restrict__ out_W,    // [U,1]
    const float* __restrict__ out_b,    // [1]
    float* __restrict__ out,            // [B,T]
    __bf16* __restrict__ ws)            // scratch
{
    const int tid  = threadIdx.x;
    const int wave = tid >> 5;
    const int ntb  = wave * 16;               // this wave's gate N-tile base
    const int b0   = blockIdx.x * TILE;       // first batch row of this tile

    __shared__ float sX[2][TILE * FEAT];      // double-buffered x_t
    __shared__ float sH[TILE * UNITS], sC[TILE * UNITS];
    __shared__ float sZ[TILE * GATES];
    __shared__ float sAttn[TILE * LEN];       // UNNORMALIZED exp-energies
    __shared__ float sE[TILE * LEN];
    __shared__ float sCtx[TILE * UNITS];
    __shared__ float sQ[TILE * ATTN];
    __shared__ float sRed[TILE * ATTN];       // softmax partials (16 rows x 10)
    __shared__ float sMax[TILE], sInv[TILE];  // row max / reciprocal row sum
    __shared__ float sWq[UNITS * ATTN], sWm[UNITS * ATTN];
    __shared__ float sBq[ATTN], sBm[ATTN], sV[ATTN], sWloc[ATTN];
    __shared__ float sEb[GATES], sDb[GATES], sOw[UNITS];
    __shared__ float sBv, sOb;

    // ---- stage small constants; wloc = Wconv @ Wd collapses to an A-vector
    for (int i = tid; i < UNITS * ATTN; i += NTHREADS) { sWq[i] = att_Wq[i]; sWm[i] = att_Wm[i]; }
    for (int i = tid; i < GATES; i += NTHREADS) { sEb[i] = enc_b[i]; sDb[i] = dec_b[i]; }
    if (tid < ATTN) {
        sBq[tid] = att_bq[tid]; sBm[tid] = att_bm[tid]; sV[tid] = att_V[tid];
        float w = 0.0f;
        for (int cf = 0; cf < CFILT; ++cf) w += loc_Wconv[cf] * loc_Wd[cf * ATTN + tid];
        sWloc[tid] = w;
    }
    if (tid < UNITS) sOw[tid] = out_W[tid];
    if (tid == 0) { sBv = att_bv[0]; sOb = out_b[0]; }
    for (int i = tid; i < TILE * UNITS; i += NTHREADS) { sH[i] = 0.0f; sC[i] = 0.0f; }
#pragma unroll
    for (int j = 0; j < XPREF; ++j) {             // stage x_0 into buffer 0
        int i = tid + j * NTHREADS;
        int m = i / FEAT, f = i % FEAT;
        sX[0][i] = inputs[((size_t)(b0 + m) * LEN + 0) * FEAT + f];
    }

    // ---- persistent weight B-fragments (loop-invariant, live in VGPRs) ----
    v16bf fWx0 = loadB_g(enc_Wx, GATES, 0,  ntb, FEAT,  GATES);
    v16bf fWx1 = loadB_g(enc_Wx, GATES, 32, ntb, FEAT,  GATES);
    v16bf fWx2 = loadB_g(enc_Wx, GATES, 64, ntb, FEAT,  GATES);
    v16bf fWhE = loadB_g(enc_Wh, GATES, 0,  ntb, UNITS, GATES);
    v16bf fWxD = loadB_g(dec_Wx, GATES, 0,  ntb, UNITS, GATES);
    v16bf fWhD = loadB_g(dec_Wh, GATES, 0,  ntb, UNITS, GATES);

    // workspace: enc_out bf16 [tile][m][u][l], mem_proj bf16 [tile][m][l][a]
    __bf16* E  = ws + (size_t)blockIdx.x * (TILE * UNITS * LEN);
    __bf16* MP = ws + (size_t)NBTILES * (TILE * UNITS * LEN)
                    + (size_t)blockIdx.x * (TILE * LEN * ATTN);

    __syncthreads();

    // ======================= encoder scan (111 steps) =======================
    for (int t = 0; t < LEN; ++t) {
        const float* cur = sX[t & 1];
        float*       nxt = sX[(t + 1) & 1];

        // prefetch x_{t+1}: loads issue now, consumed after the WMMAs
        float xr[XPREF];
        if (t + 1 < LEN) {
#pragma unroll
            for (int j = 0; j < XPREF; ++j) {
                int i = tid + j * NTHREADS;
                int m = i / FEAT, f = i % FEAT;
                xr[j] = inputs[((size_t)(b0 + m) * LEN + (t + 1)) * FEAT + f];
            }
        }

        // z[16 x 80] = x_t @ Wx + h @ Wh   (per-wave 16-gate N-tile)
        v8f  zc = {};
        v16bf af;
        af = loadA_s(cur, FEAT, 0,  FEAT);  zc = wmma_bf16(af, fWx0, zc);
        af = loadA_s(cur, FEAT, 32, FEAT);  zc = wmma_bf16(af, fWx1, zc);
        af = loadA_s(cur, FEAT, 64, FEAT);  zc = wmma_bf16(af, fWx2, zc);
        af = loadA_s(sH, UNITS, 0, UNITS);  zc = wmma_bf16(af, fWhE, zc);
        {
            int ln = threadIdx.x & 31, n = ln & 15, mh = (ln >> 4) * 8;
#pragma unroll
            for (int r = 0; r < 8; ++r) sZ[(mh + r) * GATES + ntb + n] = zc[r];
        }
        if (t + 1 < LEN) {
#pragma unroll
            for (int j = 0; j < XPREF; ++j) nxt[tid + j * NTHREADS] = xr[j];
        }
        __syncthreads();

        // LSTM pointwise (Keras gate order i,f,g,o) + stash enc_out bf16
        for (int i = tid; i < TILE * UNITS; i += NTHREADS) {
            int m = i / UNITS, u = i % UNITS;
            float zi = sZ[m * GATES +      u] + sEb[u];
            float zf = sZ[m * GATES + 20 + u] + sEb[20 + u];
            float zg = sZ[m * GATES + 40 + u] + sEb[40 + u];
            float zo = sZ[m * GATES + 60 + u] + sEb[60 + u];
            float c  = sigf(zf) * sC[i] + sigf(zi) * tanh_fast(zg);
            float h  = sigf(zo) * tanh_fast(c);
            sC[i] = c; sH[i] = h;
            E[(m * UNITS + u) * LEN + t] = (__bf16)h;
        }
        __syncthreads();

        // hoisted memory projection: mem_proj[:,t,:] = h @ Wm + bm
        for (int i = tid; i < TILE * ATTN; i += NTHREADS) {
            int m = i / ATTN, a = i % ATTN;
            float acc = sBm[a];
            for (int u = 0; u < UNITS; ++u) acc += sH[m * UNITS + u] * sWm[u * ATTN + a];
            MP[(m * LEN + t) * ATTN + a] = (__bf16)acc;
        }
        // no barrier needed: next iteration's first write is ordered by sync1
    }

    __syncthreads();
    for (int i = tid; i < TILE * UNITS; i += NTHREADS) { sH[i] = 0.0f; sC[i] = 0.0f; }
    for (int i = tid; i < TILE * LEN;   i += NTHREADS) { sAttn[i] = 0.0f; }
    if (tid < TILE) sInv[tid] = 1.0f;   // attn_prev = sAttn * sInv = 0 at t=0
    __syncthreads();

    const int srow = tid / ATTN;        // softmax row (16 rows x 10 workers)
    const int ssub = tid % ATTN;

    // ======================= decoder scan (222 steps) =======================
    for (int t = 0; t < TSTEPS; ++t) {
        // query from CELL state (faithful to reference's (c,h) unpack quirk)
        {
            int m = srow, a = ssub;
            float acc = sBq[a];
            for (int u = 0; u < UNITS; ++u) acc += sC[m * UNITS + u] * sWq[u * ATTN + a];
            sQ[tid] = acc;
        }
        __syncthreads();

        // energies e = tanh(mem_proj + q + tanh(attn*wloc)) @ V + bv,
        // row-aligned so each worker also produces a running row-max partial
        {
            int m = srow;
            float inv = sInv[m];
            float pmax = -3.0e38f;
            for (int l = ssub; l < LEN; l += ATTN) {
                float ap  = sAttn[m * LEN + l] * inv;     // normalized prev attn
                float acc = sBv;
                const __bf16* mp = &MP[(m * LEN + l) * ATTN];
#pragma unroll
                for (int a = 0; a < ATTN; ++a) {
                    float loc = tanh_fast(ap * sWloc[a]);
                    acc += sV[a] * tanh_fast((float)mp[a] + sQ[m * ATTN + a] + loc);
                }
                sE[m * LEN + l] = acc;
                pmax = fmaxf(pmax, acc);
            }
            sRed[tid] = pmax;
        }
        __syncthreads();
        if (tid < TILE) {                                  // reduce row max
            float mx = sRed[tid * ATTN];
#pragma unroll
            for (int s = 1; s < ATTN; ++s) mx = fmaxf(mx, sRed[tid * ATTN + s]);
            sMax[tid] = mx;
        }
        __syncthreads();
        {                                                  // exp + partial sums
            int m = srow;
            float mx = sMax[m], psum = 0.0f;
            for (int l = ssub; l < LEN; l += ATTN) {
                float e = __expf(sE[m * LEN + l] - mx);
                sAttn[m * LEN + l] = e;                    // unnormalized attn
                psum += e;
            }
            sRed[tid] = psum;
        }
        __syncthreads();
        if (tid < TILE) {                                  // reciprocal row sum
            float sum = sRed[tid * ATTN];
#pragma unroll
            for (int s = 1; s < ATTN; ++s) sum += sRed[tid * ATTN + s];
            sInv[tid] = __builtin_amdgcn_rcpf(sum);
        }
        __syncthreads();

        // context: per-row weighted sum (batched matvec); normalization folded
        for (int i = tid; i < TILE * UNITS; i += NTHREADS) {
            int m = i / UNITS, u = i % UNITS;
            const __bf16* ep = &E[(m * UNITS + u) * LEN];
            const float*  ap = &sAttn[m * LEN];
            float acc = 0.0f;
            for (int l = 0; l < LEN; ++l) acc += ap[l] * (float)ep[l];
            sCtx[i] = acc * sInv[m];
        }
        __syncthreads();

        // z = ctx @ dec_Wx + h @ dec_Wh
        v8f  zc = {};
        v16bf af;
        af = loadA_s(sCtx, UNITS, 0, UNITS); zc = wmma_bf16(af, fWxD, zc);
        af = loadA_s(sH,   UNITS, 0, UNITS); zc = wmma_bf16(af, fWhD, zc);
        {
            int ln = threadIdx.x & 31, n = ln & 15, mh = (ln >> 4) * 8;
#pragma unroll
            for (int r = 0; r < 8; ++r) sZ[(mh + r) * GATES + ntb + n] = zc[r];
        }
        __syncthreads();

        for (int i = tid; i < TILE * UNITS; i += NTHREADS) {
            int m = i / UNITS, u = i % UNITS;
            float zi = sZ[m * GATES +      u] + sDb[u];
            float zf = sZ[m * GATES + 20 + u] + sDb[20 + u];
            float zg = sZ[m * GATES + 40 + u] + sDb[40 + u];
            float zo = sZ[m * GATES + 60 + u] + sDb[60 + u];
            float c  = sigf(zf) * sC[i] + sigf(zi) * tanh_fast(zg);
            float h  = sigf(zo) * tanh_fast(c);
            sC[i] = c; sH[i] = h;
        }
        __syncthreads();

        // y = tanh(h @ out_W + out_b); next-step q only reads sC
        if (tid < TILE) {
            int m = tid;
            float acc = sOb;
            for (int u = 0; u < UNITS; ++u) acc += sH[m * UNITS + u] * sOw[u];
            out[(size_t)(b0 + m) * TSTEPS + t] = tanh_fast(acc);
        }
        __syncthreads();
    }
}

extern "C" void kernel_launch(void* const* d_in, const int* in_sizes, int n_in,
                              void* d_out, int out_size, void* d_ws, size_t ws_size,
                              hipStream_t stream) {
    (void)in_sizes; (void)n_in; (void)out_size; (void)ws_size;
    const float* inputs    = (const float*)d_in[0];
    const float* enc_Wx    = (const float*)d_in[1];
    const float* enc_Wh    = (const float*)d_in[2];
    const float* enc_b     = (const float*)d_in[3];
    const float* att_Wq    = (const float*)d_in[4];
    const float* att_bq    = (const float*)d_in[5];
    const float* att_Wm    = (const float*)d_in[6];
    const float* att_bm    = (const float*)d_in[7];
    const float* att_V     = (const float*)d_in[8];
    const float* att_bv    = (const float*)d_in[9];
    const float* loc_Wconv = (const float*)d_in[10];
    const float* loc_Wd    = (const float*)d_in[11];
    const float* dec_Wx    = (const float*)d_in[12];
    const float* dec_Wh    = (const float*)d_in[13];
    const float* dec_b     = (const float*)d_in[14];
    const float* out_W     = (const float*)d_in[15];
    const float* out_b     = (const float*)d_in[16];

    dim3 grid(NBTILES), block(NTHREADS);
    taco_kernel<<<grid, block, 0, stream>>>(inputs, enc_Wx, enc_Wh, enc_b,
                                            att_Wq, att_bq, att_Wm, att_bm,
                                            att_V, att_bv, loc_Wconv, loc_Wd,
                                            dec_Wx, dec_Wh, dec_b, out_W, out_b,
                                            (float*)d_out, (__bf16*)d_ws);
}
